// RNN_architecture_45595372814876
// MI455X (gfx1250) — matrix operand order repaired and checked by reading, not compiled, not run on hardware
//
#include <hip/hip_runtime.h>

typedef __attribute__((ext_vector_type(16))) _Float16 v16h;
typedef __attribute__((ext_vector_type(8)))  float    v8f;

namespace {
constexpr int kT    = 187;
constexpr int kH    = 256;
constexpr int kD1   = 64;
constexpr int kC    = 5;
constexpr int kWaves = 16;
constexpr int kTPB  = kWaves * 32;      // 512 threads = 16 wave32
constexpr int kHPad = kH + 8;           // h row stride (halves): 528B, 16B-aligned, bank-spread
constexpr int kXPad = kT + 5;           // x row stride (floats)
}

union V16H { v16h v; _Float16 e[16]; uint4 u[2]; };

__launch_bounds__(kTPB)
__global__ void rnn_fused_kernel(const float* __restrict__ x,
                                 const float* __restrict__ Wx,
                                 const float* __restrict__ Wh,
                                 const float* __restrict__ b_rnn,
                                 const float* __restrict__ W1,
                                 const float* __restrict__ b1,
                                 const float* __restrict__ W2,
                                 const float* __restrict__ b2,
                                 float* __restrict__ out)
{
    __shared__ alignas(16) _Float16 hbuf[2][16 * kHPad];  // ping-pong h state (f16)
    __shared__ float xtile[16 * kXPad];                   // staged x tile
    __shared__ float hid[16 * kD1];                       // dense1 activations

    const int tid  = threadIdx.x;
    const int wave = tid >> 5;
    const int lane = tid & 31;
    const int lhi  = lane >> 4;       // 0: lanes 0-15, 1: lanes 16-31
    const int lno  = lane & 15;
    const int batch0 = blockIdx.x * 16;

    // ---- stage x[batch0..+16)[0..T) into LDS ----
    for (int idx = tid; idx < 16 * kT; idx += kTPB) {
        const int r = idx / kT, t = idx - r * kT;
        xtile[r * kXPad + t] = x[(size_t)(batch0 + r) * kT + t];
    }

    const int ncol  = wave * 16 + lno;        // h column owned by this lane
    const float wx_n = Wx[ncol];
    const float br_n = b_rnn[ncol];

    // ---- Wh B-operands in registers: bw[kk] = Wh[kk*32..+32)[wave*16..+16) as f16 ----
    // B layout: lane holds N = lane%16; half i -> K = (lane/16)*16 + i
    V16H bw[8];
    #pragma unroll
    for (int kk = 0; kk < 8; ++kk) {
        const int kb = kk * 32 + lhi * 16;
        #pragma unroll
        for (int i = 0; i < 16; ++i)
            bw[kk].e[i] = (_Float16)Wh[(size_t)(kb + i) * kH + ncol];
    }

    // ---- h0 = 0 ----
    for (int idx = tid; idx < 16 * kHPad; idx += kTPB)
        hbuf[0][idx] = (_Float16)0.f;
    __syncthreads();

    // ---- recurrence: h = tanh(x_t*Wx + b + h@Wh), one barrier per step ----
    int p = 0;
    for (int t = 0; t < kT; ++t) {
        v8f acc = {};
        const _Float16* hb = hbuf[p];
        #pragma unroll
        for (int kk = 0; kk < 8; ++kk) {
            // A layout: lane holds M = lane%16; half i -> K = k0 + i, k0 = kk*32 + (lane/16)*16
            const int k0 = kk * 32 + lhi * 16;
            V16H a;
            a.u[0] = *(const uint4*)(hb + lno * kHPad + k0);
            a.u[1] = *(const uint4*)(hb + lno * kHPad + k0 + 8);
            acc = __builtin_amdgcn_wmma_f32_16x16x32_f16(
                false, a.v, false, bw[kk].v, (short)0, acc, false, false);
        }
        _Float16* hn = hbuf[p ^ 1];
        #pragma unroll
        for (int r = 0; r < 8; ++r) {
            const int m = r + lhi * 8;                    // C layout: M = r + 8*(lane/16)
            const float pre = acc[r] + xtile[m * kXPad + t] * wx_n + br_n;
            // tanh(v) = (1 - e^{-2v}) / (1 + e^{-2v}),  e^{-2v} = exp2(v * -2/ln2... * log2e)
            const float e2 = __builtin_amdgcn_exp2f(pre * -2.8853900817779268f);
            const float th = (1.f - e2) * __builtin_amdgcn_rcpf(1.f + e2);
            hn[m * kHPad + ncol] = (_Float16)th;
        }
        __syncthreads();
        p ^= 1;
    }

    // ---- hidden = relu(h_last @ W1 + b1): waves 0..3, WMMA ----
    if (wave < 4) {
        const int n1 = wave * 16 + lno;                   // 0..63
        V16H bw1[8];
        #pragma unroll
        for (int kk = 0; kk < 8; ++kk) {
            const int kb = kk * 32 + lhi * 16;
            #pragma unroll
            for (int i = 0; i < 16; ++i)
                bw1[kk].e[i] = (_Float16)W1[(size_t)(kb + i) * kD1 + n1];
        }
        v8f acc = {};
        const _Float16* hb = hbuf[p];
        #pragma unroll
        for (int kk = 0; kk < 8; ++kk) {
            const int k0 = kk * 32 + lhi * 16;
            V16H a;
            a.u[0] = *(const uint4*)(hb + lno * kHPad + k0);
            a.u[1] = *(const uint4*)(hb + lno * kHPad + k0 + 8);
            acc = __builtin_amdgcn_wmma_f32_16x16x32_f16(
                false, a.v, false, bw1[kk].v, (short)0, acc, false, false);
        }
        const float bias = b1[n1];
        #pragma unroll
        for (int r = 0; r < 8; ++r) {
            const int m = r + lhi * 8;
            const float v = acc[r] + bias;
            hid[m * kD1 + n1] = v > 0.f ? v : 0.f;
        }
    }
    __syncthreads();

    // ---- out = softmax(hidden @ W2 + b2): wave 0, one row per lane ----
    if (wave == 0 && lane < 16) {
        float logit[kC];
        #pragma unroll
        for (int c = 0; c < kC; ++c) logit[c] = b2[c];
        for (int k = 0; k < kD1; ++k) {
            const float hv = hid[lane * kD1 + k];
            #pragma unroll
            for (int c = 0; c < kC; ++c)
                logit[c] += hv * W2[k * kC + c];
        }
        float mx = logit[0];
        #pragma unroll
        for (int c = 1; c < kC; ++c) mx = fmaxf(mx, logit[c]);
        float ex[kC];
        float sum = 0.f;
        #pragma unroll
        for (int c = 0; c < kC; ++c) {
            ex[c] = __builtin_amdgcn_exp2f((logit[c] - mx) * 1.4426950408889634f);
            sum += ex[c];
        }
        const float rs = __builtin_amdgcn_rcpf(sum);
        #pragma unroll
        for (int c = 0; c < kC; ++c)
            out[(size_t)(batch0 + lane) * kC + c] = ex[c] * rs;
    }
}

extern "C" void kernel_launch(void* const* d_in, const int* in_sizes, int n_in,
                              void* d_out, int out_size, void* d_ws, size_t ws_size,
                              hipStream_t stream)
{
    const float* x     = (const float*)d_in[0];
    const float* Wx    = (const float*)d_in[1];
    const float* Wh    = (const float*)d_in[2];
    const float* b_rnn = (const float*)d_in[3];
    const float* W1    = (const float*)d_in[4];
    const float* b1    = (const float*)d_in[5];
    const float* W2    = (const float*)d_in[6];
    const float* b2    = (const float*)d_in[7];
    float* out = (float*)d_out;

    const int nblocks = 32768 / 16;   // 2048 batch tiles of 16 rows
    rnn_fused_kernel<<<nblocks, kTPB, 0, stream>>>(x, Wx, Wh, b_rnn, W1, b1, W2, b2, out);

    (void)in_sizes; (void)n_in; (void)out_size; (void)d_ws; (void)ws_size;
}
